// TimeInvariantNeuralODEDiffeomorphism_44349832298768
// MI455X (gfx1250) — compile-verified, matrix operand order checked by reading, not run on hardware
//
#include <hip/hip_runtime.h>
#include <stdint.h>

// ---------------------------------------------------------------------------
// TimeInvariantNeuralODEDiffeomorphism: RK4(8 steps) of z' = MLP(z)
// B=4096, DIM=256, H=1024.  Compute-bound => bf16 WMMA (f32 accum) GEMMs.
// ---------------------------------------------------------------------------

#define BB   4096
#define DIMD 256
#define HH   1024

typedef __attribute__((ext_vector_type(16))) __bf16 v16bf;
typedef __attribute__((ext_vector_type(8)))  float  v8f;

// ---------------------------------------------------------------------------
// WMMA GEMM:  C(MxN) = act( A(MxK,bf16,row-major) * Bt(NxK,bf16,row-major)^T + bias )
//
// Block = WAVES_M x WAVES_N waves; each wave computes an (MT*16) x (NT*16)
// sub-tile as MT x NT independent 16x16x32 WMMA accumulators.
// Block tile = (WAVES_M*MT*16) x (WAVES_N*NT*16).
// ---------------------------------------------------------------------------
template <int KDIM, int NDIM, int WAVES_M, int WAVES_N, int MT, int NT,
          bool SILU, bool OUT_BF16>
__global__ __launch_bounds__(WAVES_M * WAVES_N * 32)
void gemm_wmma_bf16(const __bf16* __restrict__ A,
                    const __bf16* __restrict__ Bt,
                    const float*  __restrict__ bias,
                    __bf16* __restrict__ Cb,
                    float*  __restrict__ Cf)
{
    constexpr int BM = WAVES_M * MT * 16;
    constexpr int BN = WAVES_N * NT * 16;

    const int tid  = threadIdx.x;
    const int lane = tid & 31;
    const int wave = tid >> 5;
    const int l16  = lane & 15;   // row (A) / col (B) within 16
    const int hi   = lane >> 4;   // K-half selector: contiguous k = 16*hi + s, same order for A and B

    const int wm    = wave % WAVES_M;
    const int wn    = wave / WAVES_M;
    const int waveM = blockIdx.x * BM + wm * (MT * 16);
    const int waveN = blockIdx.y * BN + wn * (NT * 16);

    // Per-lane fragment base pointers (32B naturally aligned: K stride is 512/2048B)
    const __bf16* aP[MT];
    const __bf16* bP[NT];
#pragma unroll
    for (int i = 0; i < MT; ++i)
        aP[i] = A + (size_t)(waveM + i * 16 + l16) * KDIM + hi * 16;
#pragma unroll
    for (int j = 0; j < NT; ++j)
        bP[j] = Bt + (size_t)(waveN + j * 16 + l16) * KDIM + hi * 16;

    v8f acc[MT][NT];
#pragma unroll
    for (int i = 0; i < MT; ++i)
#pragma unroll
        for (int j = 0; j < NT; ++j)
#pragma unroll
            for (int r = 0; r < 8; ++r) acc[i][j][r] = 0.0f;

#pragma unroll 2
    for (int k = 0; k < KDIM; k += 32) {
        if (k + 32 < KDIM) {
            __builtin_prefetch(aP[0] + k + 32, 0, 3);
            __builtin_prefetch(bP[0] + k + 32, 0, 3);
        }
        v16bf a[MT];
        v16bf b[NT];
#pragma unroll
        for (int i = 0; i < MT; ++i) a[i] = *(const v16bf*)(aP[i] + k);
#pragma unroll
        for (int j = 0; j < NT; ++j) b[j] = *(const v16bf*)(bP[j] + k);

#pragma unroll
        for (int i = 0; i < MT; ++i)
#pragma unroll
            for (int j = 0; j < NT; ++j)
                acc[i][j] = __builtin_amdgcn_wmma_f32_16x16x32_bf16(
                    false, a[i], false, b[j], (short)0, acc[i][j], false, false);
    }

    // Epilogue. C/D layout: VGPR r, lanes 0-15 -> M=r, lanes 16-31 -> M=8+r; N=lane&15.
#pragma unroll
    for (int i = 0; i < MT; ++i) {
#pragma unroll
        for (int j = 0; j < NT; ++j) {
            const int nIdx  = waveN + j * 16 + l16;
            const int mBase = waveM + i * 16 + hi * 8;
            const float bv  = bias[nIdx];
#pragma unroll
            for (int r = 0; r < 8; ++r) {
                float v = acc[i][j][r] + bv;
                if (SILU) {
                    v = v * (1.0f / (1.0f + __expf(-v)));   // silu
                }
                const size_t off = (size_t)(mBase + r) * NDIM + nIdx;
                if (OUT_BF16) Cb[off] = (__bf16)v;
                else          Cf[off] = v;
            }
        }
    }
}

// ---------------------------------------------------------------------------
// Elementwise helpers
// ---------------------------------------------------------------------------

// z = x - mu (mu broadcast over rows); also emit bf16 copy for first GEMM.
__global__ void prep_z_kernel(const float* __restrict__ x, const float* __restrict__ mu,
                              float* __restrict__ z, __bf16* __restrict__ zb, int total)
{
    int i = blockIdx.x * blockDim.x + threadIdx.x;
    if (i >= total) return;
    float v = x[i] - mu[i & (DIMD - 1)];
    z[i]  = v;
    zb[i] = (__bf16)v;
}

// Wt[n*K + k] = bf16(W[k*N + n])   (transpose so B-fragments are contiguous)
__global__ void convert_wt_kernel(const float* __restrict__ W, __bf16* __restrict__ Wt,
                                  int K, int N)
{
    int idx = blockIdx.x * blockDim.x + threadIdx.x;
    if (idx >= K * N) return;
    int k = idx / N;
    int n = idx - k * N;
    Wt[(size_t)n * K + k] = (__bf16)W[idx];   // coalesced read; writes combine in L2
}

// kacc = (init? acoef*kcur : kacc + acoef*kcur); zb = bf16(z + zcoef*kcur)
__global__ void rk_stage_kernel(const float* __restrict__ z, const float* __restrict__ kcur,
                                float* __restrict__ kacc, __bf16* __restrict__ zb,
                                float zcoef, float acoef, int init, int total)
{
    int i = blockIdx.x * blockDim.x + threadIdx.x;
    if (i >= total) return;
    float kc = kcur[i];
    kacc[i]  = init ? (acoef * kc) : fmaf(acoef, kc, kacc[i]);
    zb[i]    = (__bf16)fmaf(zcoef, kc, z[i]);
}

// z += dt/6 * (kacc + kcur); zb = bf16(z)
__global__ void rk_final_kernel(float* __restrict__ z, const float* __restrict__ kcur,
                                const float* __restrict__ kacc, __bf16* __restrict__ zb,
                                float dt6, int total)
{
    int i = blockIdx.x * blockDim.x + threadIdx.x;
    if (i >= total) return;
    float v = fmaf(dt6, kacc[i] + kcur[i], z[i]);
    z[i]  = v;
    zb[i] = (__bf16)v;
}

// ---------------------------------------------------------------------------
// Host orchestration
// ---------------------------------------------------------------------------
static inline void launch_vf(const __bf16* zb, __bf16* h1, __bf16* h2, float* kout,
                             const __bf16* W1t, const float* b1,
                             const __bf16* W2t, const float* b2,
                             const __bf16* W3t, const float* b3,
                             hipStream_t stream)
{
    // gemm1: (4096x256)@(256x1024)+b1, SiLU -> h1 bf16.
    //   128x128 tile, 8 waves (4x2), wave = 2x4 tiles. Grid 32x8 = 256 WGs.
    gemm_wmma_bf16<DIMD, HH, 4, 2, 2, 4, true, true>
        <<<dim3(BB / 128, HH / 128), dim3(256), 0, stream>>>(zb, W1t, b1, h1, nullptr);
    // gemm2: (4096x1024)@(1024x1024)+b2, SiLU -> h2 bf16. Same config, 256 WGs.
    gemm_wmma_bf16<HH, HH, 4, 2, 2, 4, true, true>
        <<<dim3(BB / 128, HH / 128), dim3(256), 0, stream>>>(h1, W2t, b2, h2, nullptr);
    // gemm3: (4096x1024)@(1024x256)+b3 -> kout f32.
    //   N is only 256, so shrink to 64x64 tile, 4 waves (2x2), wave = 2x2 tiles.
    //   Grid 64x4 = 256 WGs (vs 64 at 128x128) -> 4x the parallelism.
    gemm_wmma_bf16<HH, DIMD, 2, 2, 2, 2, false, false>
        <<<dim3(BB / 64, DIMD / 64), dim3(128), 0, stream>>>(h2, W3t, b3, nullptr, kout);
}

extern "C" void kernel_launch(void* const* d_in, const int* in_sizes, int n_in,
                              void* d_out, int out_size, void* d_ws, size_t ws_size,
                              hipStream_t stream)
{
    const float* x  = (const float*)d_in[0];
    const float* W1 = (const float*)d_in[1];
    const float* b1 = (const float*)d_in[2];
    const float* W2 = (const float*)d_in[3];
    const float* b2 = (const float*)d_in[4];
    const float* W3 = (const float*)d_in[5];
    const float* b3 = (const float*)d_in[6];
    const float* mu = (const float*)d_in[7];

    float* z = (float*)d_out;                     // (B, DIM) fp32 state lives in d_out

    // Workspace carve-out (256B aligned slices)
    uintptr_t p = (uintptr_t)d_ws;
    auto carve = [&](size_t bytes) -> void* {
        void* r = (void*)p;
        p += (bytes + 255) & ~(size_t)255;
        return r;
    };
    __bf16* zb   = (__bf16*)carve((size_t)BB * DIMD * 2);
    __bf16* h1   = (__bf16*)carve((size_t)BB * HH   * 2);
    __bf16* h2   = (__bf16*)carve((size_t)BB * HH   * 2);
    float*  kcur = (float*) carve((size_t)BB * DIMD * 4);
    float*  kacc = (float*) carve((size_t)BB * DIMD * 4);
    __bf16* W1t  = (__bf16*)carve((size_t)DIMD * HH * 2);
    __bf16* W2t  = (__bf16*)carve((size_t)HH * HH   * 2);
    __bf16* W3t  = (__bf16*)carve((size_t)HH * DIMD * 2);
    (void)ws_size; (void)in_sizes; (void)n_in; (void)out_size;

    const int nz = BB * DIMD;
    const dim3 eb(256);

    // Weight conversion (bf16, transposed) — cheap, once per call
    convert_wt_kernel<<<(DIMD * HH + 255) / 256, eb, 0, stream>>>(W1, W1t, DIMD, HH);
    convert_wt_kernel<<<(HH * HH   + 255) / 256, eb, 0, stream>>>(W2, W2t, HH,   HH);
    convert_wt_kernel<<<(HH * DIMD + 255) / 256, eb, 0, stream>>>(W3, W3t, HH, DIMD);

    // z0 = x - mu
    prep_z_kernel<<<(nz + 255) / 256, eb, 0, stream>>>(x, mu, z, zb, nz);

    const float dt = 0.125f;
    for (int s = 0; s < 8; ++s) {
        // k1
        launch_vf(zb, h1, h2, kcur, W1t, b1, W2t, b2, W3t, b3, stream);
        rk_stage_kernel<<<(nz + 255) / 256, eb, 0, stream>>>(z, kcur, kacc, zb, 0.5f * dt, 1.0f, 1, nz);
        // k2
        launch_vf(zb, h1, h2, kcur, W1t, b1, W2t, b2, W3t, b3, stream);
        rk_stage_kernel<<<(nz + 255) / 256, eb, 0, stream>>>(z, kcur, kacc, zb, 0.5f * dt, 2.0f, 0, nz);
        // k3
        launch_vf(zb, h1, h2, kcur, W1t, b1, W2t, b2, W3t, b3, stream);
        rk_stage_kernel<<<(nz + 255) / 256, eb, 0, stream>>>(z, kcur, kacc, zb, dt, 2.0f, 0, nz);
        // k4 + combine
        launch_vf(zb, h1, h2, kcur, W1t, b1, W2t, b2, W3t, b3, stream);
        rk_final_kernel<<<(nz + 255) / 256, eb, 0, stream>>>(z, kcur, kacc, zb, dt / 6.0f, nz);
    }
}
